// MultiHeadAttention_20710332301643
// MI455X (gfx1250) — matrix operand order repaired and checked by reading, not compiled, not run on hardware
//
#include <hip/hip_runtime.h>
#include <hip/hip_bf16.h>

// ---------------------------------------------------------------------------
// MI455X (gfx1250) packed-varlen multi-head attention.
// fp32 -> f16 once; all GEMMs + QK^T + PV via v_wmma_f32_16x16x32_f16 (f32
// accumulate); softmax fp32.  GEMM tiles fed by the Tensor Data Mover
// (tensor_load_to_lds + s_wait_tensorcnt) with double-buffered LDS.
// ---------------------------------------------------------------------------

typedef __attribute__((ext_vector_type(16))) _Float16 v16h;
typedef __attribute__((ext_vector_type(8)))  _Float16 v8h;
typedef __attribute__((ext_vector_type(8)))  float    v8f;
typedef __attribute__((ext_vector_type(4)))  unsigned int u32x4;
typedef __attribute__((ext_vector_type(8)))  int i32x8;
typedef __attribute__((ext_vector_type(4)))  int i32x4;

union HH { v16h v16; v8h h8[2]; };

// --- TDM availability / arity detection (compile-time) ---------------------
#if defined(__HIP_DEVICE_COMPILE__) && defined(__has_builtin)
#if __has_builtin(__builtin_amdgcn_tensor_load_to_lds)
#define USE_TDM 1
#endif
#endif
#ifndef USE_TDM
#define USE_TDM 0
#endif

#if __has_include(<hip/amd_detail/amd_gfx1250_TDM.h>)
#define TDM_6ARG 1
#else
#define TDM_6ARG 0
#endif

__device__ inline v8f wmma_f16f32(v16h a, v16h b, v8f c) {
  return __builtin_amdgcn_wmma_f32_16x16x32_f16(
      /*neg_a=*/false, a, /*neg_b=*/false, b,
      /*c_mod=*/(short)0, c, /*reuse_a=*/false, /*reuse_b=*/false);
}

__device__ inline v8f zero8f() {
  v8f z;
#pragma unroll
  for (int i = 0; i < 8; ++i) z[i] = 0.0f;
  return z;
}
__device__ inline v8h zero8h() {
  v8h z;
#pragma unroll
  for (int i = 0; i < 8; ++i) z[i] = (_Float16)0.0f;
  return z;
}

#if USE_TDM
// Issue one 2D TDM load: tile [tile1 rows x tile0 elems] of f16 starting at
// gptr (row stride = stride0 elems) into LDS at lds_off, with hardware LDS
// padding of 4 DWORDs every 16 DWORDs (row stride 32 f16 -> 40 f16 in LDS).
// D# per CDNA5 ISA ch.8: group0 = {count, lds_addr, global_addr, type=2},
// group1 = {data_size=2B, pad cfg, tensor dims, tile dims, dim0 stride}.
__device__ inline void tdm_load_2d(unsigned lds_off, const _Float16* gptr,
                                   unsigned dim0, unsigned dim1,
                                   unsigned tile0, unsigned tile1,
                                   unsigned long long stride0) {
  unsigned long long ga = (unsigned long long)(size_t)gptr;
  u32x4 g0;
  g0[0] = 1u;                                           // count=1, user D#
  g0[1] = lds_off;                                      // lds_addr (bytes)
  g0[2] = (unsigned)(ga & 0xffffffffull);               // global_addr[31:0]
  g0[3] = (unsigned)((ga >> 32) & 0x01fffffful)         // global_addr[56:32]
        | 0x80000000u;                                  // type = 2 ("image")
  unsigned d[8];
  d[0] = (1u << 16)            // data_size = 1 (2 bytes)
       | (1u << 20)            // pad_enable
       | (3u << 22)            // pad_interval: 16 DWORDs
       | (3u << 25);           // pad_amount:   4 DWORDs
  d[1] = (dim0 & 0xffffu) << 16;                        // tensor_dim0[15:0]
  d[2] = (dim0 >> 16) | ((dim1 & 0xffffu) << 16);       // dim0[31:16]|dim1[15:0]
  d[3] = (dim1 >> 16) | ((tile0 & 0xffffu) << 16);      // dim1[31:16]|tile_dim0
  d[4] = tile1 & 0xffffu;                               // tile_dim1 (tile_dim2=0)
  d[5] = (unsigned)(stride0 & 0xffffffffull);           // tensor_dim0_stride lo
  d[6] = (unsigned)((stride0 >> 32) & 0xffffull);       // stride hi (dim1_stride=0)
  d[7] = 0;
  i32x8 g1;
#pragma unroll
  for (int i = 0; i < 8; ++i) g1[i] = (int)d[i];
  i32x4 z4;
#pragma unroll
  for (int i = 0; i < 4; ++i) z4[i] = 0;
#if TDM_6ARG
  i32x8 z8;
#pragma unroll
  for (int i = 0; i < 8; ++i) z8[i] = 0;
  __builtin_amdgcn_tensor_load_to_lds(g0, g1, z4, z4, z8, 0);
#else
  __builtin_amdgcn_tensor_load_to_lds(g0, g1, z4, z4, 0);
#endif
}

__device__ inline void wait_tensor0() {
#if __has_builtin(__builtin_amdgcn_s_wait_tensorcnt)
  __builtin_amdgcn_s_wait_tensorcnt(0);
#else
  asm volatile("s_wait_tensorcnt 0x0" ::: "memory");
#endif
}
#endif  // USE_TDM

// ---------------------------------------------------------------------------
// fp32 -> f16 conversion
// ---------------------------------------------------------------------------
__global__ void __launch_bounds__(256) cvt_f32_to_f16(const float* __restrict__ src,
                                                      _Float16* __restrict__ dst, int n) {
  int i = blockIdx.x * 256 + threadIdx.x;
  if (i < n) dst[i] = (_Float16)src[i];
}

// ---------------------------------------------------------------------------
// GEMM: C[M,1024] = A[M,1024] @ W[1024,1024]^T (+ bias)
// Block tile 128x128, 256 threads = 8 waves (2x4), wave tile 64x32 (4x2 WMMA).
// TDM path: wave 0 DMAs next K-step's A/B tiles into the spare LDS buffer
// while all waves run WMMAs on the current one; one barrier per K-step.
// ---------------------------------------------------------------------------
template <typename OutT>
__global__ void __launch_bounds__(256) gemm_xwT(const _Float16* __restrict__ A,
                                                const _Float16* __restrict__ W,
                                                const float* __restrict__ bias,
                                                OutT* __restrict__ C, int M) {
  constexpr int KD = 1024;
  constexpr int LDA = 40;  // f16 row stride in LDS (32 data + 8 pad = 80B)
  __shared__ _Float16 sA[2][128 * LDA];
  __shared__ _Float16 sB[2][128 * LDA];

  const int tid  = threadIdx.x;
  const int lane = tid & 31;
  const int w    = tid >> 5;
  const int hi   = lane >> 4;
  const int lo   = lane & 15;
  const int mbase = blockIdx.x * 128;
  const int nbase = blockIdx.y * 128;
  const int mw = (w >> 2) * 64;   // wave row offset inside block (0 or 64)
  const int nw = (w & 3) * 32;    // wave col offset inside block (0..96)

  v8f acc[4][2];
#pragma unroll
  for (int mi = 0; mi < 4; ++mi)
#pragma unroll
    for (int ni = 0; ni < 2; ++ni) acc[mi][ni] = zero8f();

#if USE_TDM
  const unsigned ldsA[2] = {(unsigned)(size_t)&sA[0][0], (unsigned)(size_t)&sA[1][0]};
  const unsigned ldsB[2] = {(unsigned)(size_t)&sB[0][0], (unsigned)(size_t)&sB[1][0]};
  const unsigned remM = (unsigned)(M - mbase);        // rows left; OOB rows -> 0
  const unsigned remN = (unsigned)(1024 - nbase);
  if (w == 0) {
    tdm_load_2d(ldsA[0], A + (size_t)mbase * KD, KD, remM, 32, 128, KD);
    tdm_load_2d(ldsB[0], W + (size_t)nbase * KD, KD, remN, 32, 128, KD);
  }
#else
  const int crow = tid >> 1;         // 0..127 : tile row this thread copies
  const int ckc  = (tid & 1) * 16;   // 0/16   : f16 chunk within 32-wide K step
#endif

  int buf = 0;
  for (int k0 = 0; k0 < KD; k0 += 32, buf ^= 1) {
#if USE_TDM
    if (w == 0) wait_tensor0();      // current buffer's DMA complete
    __syncthreads();                 // publish LDS; spare buffer free again
    if (w == 0 && k0 + 32 < KD) {
      tdm_load_2d(ldsA[buf ^ 1], A + (size_t)mbase * KD + k0 + 32,
                  KD - (k0 + 32), remM, 32, 128, KD);
      tdm_load_2d(ldsB[buf ^ 1], W + (size_t)nbase * KD + k0 + 32,
                  KD - (k0 + 32), remN, 32, 128, KD);
    }
#else
    __syncthreads();
    {  // stage A tile [128 x 32]
      v8h a0 = zero8h(), a1 = zero8h();
      const int gr = mbase + crow;
      if (gr < M) {
        const v8h* p = (const v8h*)(A + (size_t)gr * KD + k0 + ckc);
        a0 = p[0]; a1 = p[1];
      }
      v8h* s = (v8h*)(sA[0] + crow * LDA + ckc);
      s[0] = a0; s[1] = a1;
    }
    {  // stage B tile: sB[n][k] = W[nbase+n][k0+k]
      const v8h* p = (const v8h*)(W + (size_t)(nbase + crow) * KD + k0 + ckc);
      v8h b0 = p[0], b1 = p[1];
      v8h* s = (v8h*)(sB[0] + crow * LDA + ckc);
      s[0] = b0; s[1] = b1;
    }
    __syncthreads();
#endif
    const _Float16* curA = sA[USE_TDM ? buf : 0];
    const _Float16* curB = sB[USE_TDM ? buf : 0];

    // fragments (contiguous 16B LDS loads matching the ISA VGPR layouts)
    HH af[4], bf[2];
#pragma unroll
    for (int mi = 0; mi < 4; ++mi) {
      const _Float16* base = curA + (mw + mi * 16 + lo) * LDA;
      af[mi].h8[0] = *(const v8h*)(base + hi * 8);        // K = hi*8 + 0..7
      af[mi].h8[1] = *(const v8h*)(base + 16 + hi * 8);   // K = 16+hi*8 + 0..7
    }
#pragma unroll
    for (int ni = 0; ni < 2; ++ni) {
      const _Float16* base = curB + (nw + ni * 16 + lo) * LDA;
      bf[ni].h8[0] = *(const v8h*)(base + hi * 8);        // K = hi*8 + 0..7
      bf[ni].h8[1] = *(const v8h*)(base + 16 + hi * 8);   // K = 16+hi*8 + 0..7
    }
#pragma unroll
    for (int mi = 0; mi < 4; ++mi)
#pragma unroll
      for (int ni = 0; ni < 2; ++ni)
        acc[mi][ni] = wmma_f16f32(af[mi].v16, bf[ni].v16, acc[mi][ni]);
  }

  // epilogue: bias add (f32) + store
#pragma unroll
  for (int mi = 0; mi < 4; ++mi)
#pragma unroll
    for (int ni = 0; ni < 2; ++ni)
#pragma unroll
      for (int v = 0; v < 8; ++v) {
        const int r = mbase + mw + mi * 16 + v + hi * 8;
        const int c = nbase + nw + ni * 16 + lo;
        if (r < M) {
          float val = acc[mi][ni][v] + (bias ? bias[c] : 0.0f);
          C[(size_t)r * KD + c] = (OutT)val;
        }
      }
}

// ---------------------------------------------------------------------------
// Flash attention over packed sequences.
// Grid: (M/64 query blocks, 16 heads).  128 threads = 4 waves, 16 queries/wave.
// 32-key steps; online softmax fp32; QK^T and PV via WMMA f16; next block's
// K/V rows prefetched (global_prefetch_b8) while current block computes.
// ---------------------------------------------------------------------------
__global__ void __launch_bounds__(128) attn_flash(const _Float16* __restrict__ Q,
                                                  const _Float16* __restrict__ K,
                                                  const _Float16* __restrict__ V,
                                                  const int* __restrict__ cu,
                                                  _Float16* __restrict__ Ctx) {
  constexpr int D = 1024, HD = 64;
  constexpr int LDK = 72;   // sK row stride (f16): 144B, 16B multiple
  constexpr int LDV = 40;   // sVt row stride
  constexpr int LDP = 40;   // per-wave P scratch stride
  __shared__ _Float16 sK[32 * LDK];        // [key][hd]
  __shared__ _Float16 sVt[64 * LDV];       // [hd][key]  (V transposed)
  __shared__ _Float16 sP[4][16 * LDP];     // per-wave P(16x32) scratch

  const int tid  = threadIdx.x;
  const int lane = tid & 31;
  const int w    = tid >> 5;
  const int hi   = lane >> 4;
  const int lo   = lane & 15;
  const int h    = blockIdx.y;
  const int tok0 = blockIdx.x * 64;

  // locate sequence (lengths are multiples of 64 -> block never straddles)
  int b = 0;
#pragma unroll
  for (int i = 1; i < 8; ++i)
    if (tok0 >= cu[i]) b = i;
  const int segstart = cu[b];
  const int L = cu[b + 1] - segstart;
  const int qtok = tok0 + w * 16;  // first query token of this wave

  // Q fragments: 16x64 tile = two 16x32 A-fragments, straight from global
  HH qf[2];
  {
    const _Float16* qrow = Q + (size_t)(qtok + lo) * D + h * HD;
#pragma unroll
    for (int ks = 0; ks < 2; ++ks) {
      qf[ks].h8[0] = *(const v8h*)(qrow + ks * 32 + hi * 8);
      qf[ks].h8[1] = *(const v8h*)(qrow + ks * 32 + 16 + hi * 8);
    }
  }

  float m_i[8], l_i[8];
  v8f o[4];
#pragma unroll
  for (int v = 0; v < 8; ++v) { m_i[v] = -1e30f; l_i[v] = 0.0f; }
#pragma unroll
  for (int nt = 0; nt < 4; ++nt) o[nt] = zero8f();

  const int ck = tid >> 2;         // 0..31 key row this thread copies
  const int cc = (tid & 3) * 16;   // hd chunk (0/16/32/48)
  const float scale = 0.125f;      // hd^-0.5

  for (int kb = 0; kb < L; kb += 32) {
    __syncthreads();
    // --- stage K tile [32 x 64] and Vt tile [64 x 32] ---
    {
      v8h k0v = zero8h(), k1v = zero8h(), v0v = zero8h(), v1v = zero8h();
      const int kt = kb + ck;
      if (kt < L) {
        const v8h* kp = (const v8h*)(K + (size_t)(segstart + kt) * D + h * HD + cc);
        k0v = kp[0]; k1v = kp[1];
        const v8h* vp = (const v8h*)(V + (size_t)(segstart + kt) * D + h * HD + cc);
        v0v = vp[0]; v1v = vp[1];
      }
      v8h* sk = (v8h*)(sK + ck * LDK + cc);
      sk[0] = k0v; sk[1] = k1v;
#pragma unroll
      for (int j = 0; j < 8; ++j) {
        sVt[(cc + j) * LDV + ck]     = v0v[j];
        sVt[(cc + 8 + j) * LDV + ck] = v1v[j];
      }
      // prefetch next key block into cache while this one computes
      const int kt2 = kb + 32 + ck;
      if (kt2 < L) {
        __builtin_prefetch(K + (size_t)(segstart + kt2) * D + h * HD + cc, 0, 1);
        __builtin_prefetch(V + (size_t)(segstart + kt2) * D + h * HD + cc, 0, 1);
      }
    }
    __syncthreads();

    // --- S = Q K^T : two 16x16 tiles, each accumulating over hd=64 ---
    v8f s[2];
#pragma unroll
    for (int nt = 0; nt < 2; ++nt) {
      v8f z = zero8f();
#pragma unroll
      for (int ks = 0; ks < 2; ++ks) {
        HH bf;
        const _Float16* kbase = sK + (nt * 16 + lo) * LDK + ks * 32 + hi * 8;
        bf.h8[0] = *(const v8h*)(kbase);
        bf.h8[1] = *(const v8h*)(kbase + 16);
        z = wmma_f16f32(qf[ks].v16, bf.v16, z);
      }
      s[nt] = z;
    }

    // --- scale + key mask + online softmax (rows span 16 lanes) ---
    float rm[8];
#pragma unroll
    for (int v = 0; v < 8; ++v) {
      float s0 = s[0][v] * scale;
      float s1 = s[1][v] * scale;
      if (kb + lo >= L)      s0 = -1e30f;
      if (kb + 16 + lo >= L) s1 = -1e30f;
      s[0][v] = s0; s[1][v] = s1;
      rm[v] = fmaxf(s0, s1);
    }
#pragma unroll
    for (int v = 0; v < 8; ++v)
#pragma unroll
      for (int off = 1; off < 16; off <<= 1)
        rm[v] = fmaxf(rm[v], __shfl_xor(rm[v], off, 32));

    float alpha[8], psum[8];
#pragma unroll
    for (int v = 0; v < 8; ++v) {
      const float mnew = fmaxf(m_i[v], rm[v]);
      alpha[v] = __expf(m_i[v] - mnew);
      m_i[v] = mnew;
      const float p0 = __expf(s[0][v] - mnew);
      const float p1 = __expf(s[1][v] - mnew);
      s[0][v] = p0; s[1][v] = p1;
      psum[v] = p0 + p1;
    }
#pragma unroll
    for (int v = 0; v < 8; ++v)
#pragma unroll
      for (int off = 1; off < 16; off <<= 1)
        psum[v] += __shfl_xor(psum[v], off, 32);
#pragma unroll
    for (int v = 0; v < 8; ++v) l_i[v] = l_i[v] * alpha[v] + psum[v];
#pragma unroll
    for (int nt = 0; nt < 4; ++nt)
#pragma unroll
      for (int v = 0; v < 8; ++v) o[nt][v] *= alpha[v];

    // --- P (C-layout) -> per-wave LDS -> A-fragment (DS in-order per wave) ---
    _Float16* pw = sP[w];
#pragma unroll
    for (int v = 0; v < 8; ++v) {
      pw[(v + hi * 8) * LDP + lo]      = (_Float16)s[0][v];
      pw[(v + hi * 8) * LDP + 16 + lo] = (_Float16)s[1][v];
    }
    __builtin_amdgcn_wave_barrier();
    HH pf;
    {
      const _Float16* pb = pw + lo * LDP;
      pf.h8[0] = *(const v8h*)(pb + hi * 8);
      pf.h8[1] = *(const v8h*)(pb + 16 + hi * 8);
    }

    // --- O += P V : four 16x16 output tiles over hd ---
#pragma unroll
    for (int nt = 0; nt < 4; ++nt) {
      HH bf;
      const _Float16* vb = sVt + (nt * 16 + lo) * LDV + hi * 8;
      bf.h8[0] = *(const v8h*)(vb);
      bf.h8[1] = *(const v8h*)(vb + 16);
      o[nt] = wmma_f16f32(pf.v16, bf.v16, o[nt]);
    }
  }

  // --- normalize and store ctx (f16, packed [tok][h*64+hd]) ---
#pragma unroll
  for (int nt = 0; nt < 4; ++nt)
#pragma unroll
    for (int v = 0; v < 8; ++v) {
      const int r = qtok + v + hi * 8;
      const float val = o[nt][v] / l_i[v];
      Ctx[(size_t)r * D + h * HD + nt * 16 + lo] = (_Float16)val;
    }
}

// ---------------------------------------------------------------------------
// Launcher
// ---------------------------------------------------------------------------
extern "C" void kernel_launch(void* const* d_in, const int* in_sizes, int n_in,
                              void* d_out, int out_size, void* d_ws, size_t ws_size,
                              hipStream_t stream) {
  constexpr int D = 1024;
  const float* x  = (const float*)d_in[0];
  const float* Wq = (const float*)d_in[1];
  const float* bq = (const float*)d_in[2];
  const float* Wk = (const float*)d_in[3];
  const float* Wv = (const float*)d_in[4];
  const float* bv = (const float*)d_in[5];
  const float* Wo = (const float*)d_in[6];
  const float* bo = (const float*)d_in[7];
  const int*   cu = (const int*)d_in[8];
  float* out = (float*)d_out;

  const int M = in_sizes[0] / D;  // 5952 (all lengths multiples of 64)

  char* ws = (char*)d_ws;
  size_t off = 0;
  auto alloc = [&](size_t bytes) -> _Float16* {
    _Float16* p = (_Float16*)(ws + off);
    off = (off + bytes + 255) & ~(size_t)255;
    return p;
  };
  const size_t actBytes = (size_t)M * D * sizeof(_Float16);
  const size_t wBytes   = (size_t)D * D * sizeof(_Float16);
  _Float16* xh   = alloc(actBytes);
  _Float16* Wqh  = alloc(wBytes);
  _Float16* Wkh  = alloc(wBytes);
  _Float16* Wvh  = alloc(wBytes);
  _Float16* Woh  = alloc(wBytes);
  _Float16* Qh   = alloc(actBytes);
  _Float16* Kh   = alloc(actBytes);
  _Float16* Vh   = alloc(actBytes);
  _Float16* Ctxh = alloc(actBytes);

  // 1) fp32 -> f16
  {
    const int nx = M * D, nw = D * D;
    cvt_f32_to_f16<<<(nx + 255) / 256, 256, 0, stream>>>(x, xh, nx);
    cvt_f32_to_f16<<<(nw + 255) / 256, 256, 0, stream>>>(Wq, Wqh, nw);
    cvt_f32_to_f16<<<(nw + 255) / 256, 256, 0, stream>>>(Wk, Wkh, nw);
    cvt_f32_to_f16<<<(nw + 255) / 256, 256, 0, stream>>>(Wv, Wvh, nw);
    cvt_f32_to_f16<<<(nw + 255) / 256, 256, 0, stream>>>(Wo, Woh, nw);
  }

  // 2) QKV projections (y = x @ W^T + b ; key has no bias)
  dim3 ggrid((M + 127) / 128, D / 128);
  gemm_xwT<_Float16><<<ggrid, 256, 0, stream>>>(xh, Wqh, bq, Qh, M);
  gemm_xwT<_Float16><<<ggrid, 256, 0, stream>>>(xh, Wkh, (const float*)nullptr, Kh, M);
  gemm_xwT<_Float16><<<ggrid, 256, 0, stream>>>(xh, Wvh, bv, Vh, M);

  // 3) flash attention per (64-query block, head)
  attn_flash<<<dim3(M / 64, 16), 128, 0, stream>>>(Qh, Kh, Vh, cu, Ctxh);

  // 4) output projection -> fp32 result
  gemm_xwT<float><<<ggrid, 256, 0, stream>>>(Ctxh, Woh, bo, out, M);
}
